// MeshAutoencoder_57887569215571
// MI455X (gfx1250) — matrix operand order repaired and checked by reading, not compiled, not run on hardware
//
#include <hip/hip_runtime.h>

// ---------------------------------------------------------------------------
// MeshAutoencoder forward for MI455X (gfx1250, wave32, WMMA).
// All GEMM-shaped work (incl. the dominant 252-GFLOP residual-VQ distance
// matmul) runs on v_wmma_f32_16x16x32_bf16 with f32 accumulation.
// Activations are kept as bf16 in memory so the GEMM k-loop is pure
// load_b128 + wmma (no per-iteration cvt VALU).
// ---------------------------------------------------------------------------

typedef __attribute__((ext_vector_type(16))) __bf16 v16bf;
typedef __attribute__((ext_vector_type(8)))  __bf16 v8bf;
typedef __attribute__((ext_vector_type(8)))  float   v8f;
typedef __attribute__((ext_vector_type(4)))  float   v4f;

#define BB    2
#define NVV   10000
#define NFF   20000
#define EEE   60000
#define DIMF  512
#define DCB   192
#define KIN   576
#define NCB   16384
#define MROWS 40000   // BB*NFF
#define VROWS 20000   // BB*NVV

static __device__ __forceinline__ __bf16 f2bf(float f) {
  unsigned u = __builtin_bit_cast(unsigned, f);
  u += 0x7FFFu + ((u >> 16) & 1u);                 // round-to-nearest-even
  unsigned short h = (unsigned short)(u >> 16);
  return __builtin_bit_cast(__bf16, h);
}

// A-fragment (16x32 bf16, MxK): lane m=lane%16 holds row m; kh=lane/16 selects
// K = {kh*8 .. kh*8+7} and {16+kh*8 .. 16+kh*8+7}  (ISA 7.12.2 table).
static __device__ __forceinline__ v16bf load_a_frag_bf16(const __bf16* __restrict__ A,
                                                         int row, int ldk, int k0, int lane) {
  const int kh = lane >> 4;
  const __bf16* p = A + (size_t)row * ldk + k0 + kh * 8;
  v8bf lo = *(const v8bf*)(p);
  v8bf hi = *(const v8bf*)(p + 16);
  v16bf a;
#pragma unroll
  for (int i = 0; i < 8; ++i) { a[i] = lo[i]; a[8 + i] = hi[i]; }
  return a;
}

// Same fragment built from an f32 source (used only outside hot loops: VQ setup).
static __device__ __forceinline__ v16bf load_a_frag_f32(const float* __restrict__ A,
                                                        int row, int ldk, int k0, int lane) {
  const int kh = lane >> 4;
  const float* p = A + (size_t)row * ldk + k0 + kh * 8;
  v4f q0 = *(const v4f*)(p);
  v4f q1 = *(const v4f*)(p + 4);
  v4f q2 = *(const v4f*)(p + 16);
  v4f q3 = *(const v4f*)(p + 20);
  v16bf a;
#pragma unroll
  for (int i = 0; i < 4; ++i) {
    a[i]      = f2bf(q0[i]);
    a[4 + i]  = f2bf(q1[i]);
    a[8 + i]  = f2bf(q2[i]);
    a[12 + i] = f2bf(q3[i]);
  }
  return a;
}

// B-fragment (32x16 bf16, KxN), stored N-major (Bt[n][k]): lane n=lane%16 is
// the column, kh=lane/16 selects K = {kh*8 .. kh*8+7} and {16+kh*8 .. 16+kh*8+7}.
static __device__ __forceinline__ v16bf load_b_frag_bf16(const __bf16* __restrict__ Bt,
                                                         int n, int ldk, int k0, int lane) {
  const int kh = lane >> 4;
  const __bf16* p = Bt + (size_t)n * ldk + k0 + kh * 8;
  v8bf lo = *(const v8bf*)(p);
  v8bf hi = *(const v8bf*)(p + 16);
  v16bf b;
#pragma unroll
  for (int i = 0; i < 8; ++i) { b[i] = lo[i]; b[8 + i] = hi[i]; }
  return b;
}

static __device__ __forceinline__ v8f wmma_bf16(v16bf a, v16bf b, v8f c) {
  return __builtin_amdgcn_wmma_f32_16x16x32_bf16(false, a, false, b, (short)0, c,
                                                 false, false);
}

// ---------------------------------------------------------------------------
// Dual-source GEMM:  C[M,N] = A1[M,K1]*B1t[N,K1] + A2[M,K2]*B2t[N,K2] + bias
// (A2 may be null).  All operands bf16, accumulation f32.  Writes f32 C and/or
// a bf16 copy.  16x64 tile per wave, 8 waves/block (32x256 block tile).
// ---------------------------------------------------------------------------
__global__ __launch_bounds__(256) void gemm_kernel(
    const __bf16* __restrict__ A1, const __bf16* __restrict__ B1, int K1,
    const __bf16* __restrict__ A2, const __bf16* __restrict__ B2, int K2,
    const float* __restrict__ bias,
    float* __restrict__ Cf, __bf16* __restrict__ Cb,
    int M, int N) {
  const int lane = threadIdx.x & 31;
  const int wave = threadIdx.x >> 5;
  const int rowBase = blockIdx.y * 32 + (wave >> 2) * 16;
  const int colBase = blockIdx.x * 256 + (wave & 3) * 64;
  const int m = lane & 15, kh = lane >> 4;
  int rowA = rowBase + m; if (rowA > M - 1) rowA = M - 1;
  int ncl[4];
#pragma unroll
  for (int t = 0; t < 4; ++t) {
    int n = colBase + t * 16 + m; if (n > N - 1) n = N - 1;
    ncl[t] = n;
  }

  v8f acc[4];
#pragma unroll
  for (int t = 0; t < 4; ++t)
#pragma unroll
    for (int i = 0; i < 8; ++i) acc[t][i] = 0.0f;

  for (int k0 = 0; k0 < K1; k0 += 32) {
    v16bf a = load_a_frag_bf16(A1, rowA, K1, k0, lane);
#pragma unroll
    for (int t = 0; t < 4; ++t) {
      v16bf b = load_b_frag_bf16(B1, ncl[t], K1, k0, lane);
      acc[t] = wmma_bf16(a, b, acc[t]);
    }
  }
  if (A2) {
    for (int k0 = 0; k0 < K2; k0 += 32) {
      v16bf a = load_a_frag_bf16(A2, rowA, K2, k0, lane);
#pragma unroll
      for (int t = 0; t < 4; ++t) {
        v16bf b = load_b_frag_bf16(B2, ncl[t], K2, k0, lane);
        acc[t] = wmma_bf16(a, b, acc[t]);
      }
    }
  }
  // C/D layout: VGPR r -> row rowBase + r + 8*kh, col = colBase + t*16 + m
#pragma unroll
  for (int t = 0; t < 4; ++t) {
    const int col = colBase + t * 16 + m;
    if (col >= N) continue;
    const float bv = bias ? bias[col] : 0.0f;
#pragma unroll
    for (int r = 0; r < 8; ++r) {
      const int row = rowBase + r + kh * 8;
      if (row < M) {
        size_t o = (size_t)row * N + col;
        float v = acc[t][r] + bv;
        if (Cf) Cf[o] = v;
        if (Cb) Cb[o] = f2bf(v);
      }
    }
  }
}

// ---------------------------------------------------------------------------
// Residual VQ: one wave per 16-row tile of avg[20000,192].  K=192 A-fragments
// live in VGPRs across the whole codebook sweep and both quantizer passes.
// argmin over d2 = cb_sq[n] - 2*dot (the ||r||^2 term is row-constant).
// ---------------------------------------------------------------------------
__global__ __launch_bounds__(32) void vq_kernel(const float* __restrict__ avg,
                                                const __bf16* __restrict__ cbb,
                                                const float* __restrict__ cbf,
                                                const float* __restrict__ cbsq,
                                                int* __restrict__ idx0,
                                                int* __restrict__ idx1) {
  __shared__ int sidx[16];
  const int lane = threadIdx.x;
  const int m = lane & 15, kh = lane >> 4;
  const int rowBase = blockIdx.x * 16;
  const int row = rowBase + m;

  v16bf afrag[6];
#pragma unroll
  for (int f = 0; f < 6; ++f) afrag[f] = load_a_frag_f32(avg, row, DCB, f * 32, lane);

  for (int iter = 0; iter < 2; ++iter) {
    float minv[8]; int mini[8];
#pragma unroll
    for (int r = 0; r < 8; ++r) { minv[r] = 3.4e38f; mini[r] = 0; }

    for (int nt = 0; nt < NCB / 16; ++nt) {
      const int n = nt * 16 + m;
      v8f c;
#pragma unroll
      for (int i = 0; i < 8; ++i) c[i] = 0.0f;
#pragma unroll
      for (int f = 0; f < 6; ++f) {
        v16bf b = load_b_frag_bf16(cbb, n, DCB, f * 32, lane);
        c = wmma_bf16(afrag[f], b, c);
      }
      const float cq = cbsq[n];
#pragma unroll
      for (int r = 0; r < 8; ++r) {
        float d2 = cq - 2.0f * c[r];
        if (d2 < minv[r]) { minv[r] = d2; mini[r] = n; }
      }
    }
    // argmin across the 16 lanes of each half (rows 0-7 in lanes 0-15,
    // rows 8-15 in lanes 16-31); xor offsets < 16 never cross halves.
#pragma unroll
    for (int off = 1; off < 16; off <<= 1) {
#pragma unroll
      for (int r = 0; r < 8; ++r) {
        float ov = __shfl_xor(minv[r], off, 32);
        int   oi = __shfl_xor(mini[r], off, 32);
        if (ov < minv[r] || (ov == minv[r] && oi < mini[r])) { minv[r] = ov; mini[r] = oi; }
      }
    }
    if (m == 0) {
#pragma unroll
      for (int r = 0; r < 8; ++r) {
        const int rr = kh * 8 + r;
        sidx[rr] = mini[r];
        (iter == 0 ? idx0 : idx1)[rowBase + rr] = mini[r];
      }
    }
    __syncthreads();
    if (iter == 0) {   // rebuild A as bf16(avg - codebook[idx]) in-register
      const int ci = sidx[m];
      const float* av = avg + (size_t)row * DCB;
      const float* cv = cbf + (size_t)ci * DCB;
#pragma unroll
      for (int f = 0; f < 6; ++f) {
        const int k0 = f * 32 + kh * 8;
#pragma unroll
        for (int i = 0; i < 8; ++i) {
          afrag[f][i]     = f2bf(av[k0 + i]      - cv[k0 + i]);
          afrag[f][8 + i] = f2bf(av[k0 + 16 + i] - cv[k0 + 16 + i]);
        }
      }
    }
    __syncthreads();
  }
}

// ------------------------- elementwise / gather kernels --------------------
__global__ void zero_kernel(float* __restrict__ p, int n) {
  int i = blockIdx.x * blockDim.x + threadIdx.x;
  if (i < n) p[i] = 0.0f;
}

__global__ void transpose_bf16_kernel(const float* __restrict__ src, __bf16* __restrict__ dst,
                                      int K, int N, int total) {
  int i = blockIdx.x * blockDim.x + threadIdx.x;
  if (i >= total) return;
  int n = i / K, k = i % K;
  dst[i] = f2bf(src[(size_t)k * N + n]);
}

__global__ void cbconv_kernel(const float* __restrict__ cbf, __bf16* __restrict__ cbb, int total) {
  int i = blockIdx.x * blockDim.x + threadIdx.x;
  if (i < total) cbb[i] = f2bf(cbf[i]);
}

__global__ void cbsq_kernel(const float* __restrict__ cbf, float* __restrict__ cbsq, int n) {
  int i = blockIdx.x * blockDim.x + threadIdx.x;
  if (i >= n) return;
  const float* p = cbf + (size_t)i * DCB;
  float s = 0.0f;
  for (int k = 0; k < DCB; ++k) s += p[k] * p[k];
  cbsq[i] = s;
}

__global__ void embed_kernel(const float* __restrict__ vertices, const int* __restrict__ faces,
                             const float* __restrict__ coor_embed, __bf16* __restrict__ emb,
                             int total) {
  int i = blockIdx.x * blockDim.x + threadIdx.x;
  if (i >= total) return;
  int d  = i & 63;
  int t  = i >> 6;          // fr*9 + c9
  int c9 = t % 9;
  int fr = t / 9;           // b*NFF + f
  int j = c9 / 3, cc = c9 % 3;
  int b = fr / NFF;
  int v = faces[(size_t)fr * 3 + j];
  float coord = vertices[((size_t)b * NVV + v) * 3 + cc];
  float x = (coord + 1.0f) * 0.5f * 128.0f - 0.5f;
  int di = (int)rintf(x);                 // RNE matches jnp.round
  di = di < 0 ? 0 : (di > 127 ? 127 : di);
  emb[i] = f2bf(coor_embed[di * 64 + d]);
}

__global__ void cnt_kernel(const int* __restrict__ fedges, float* __restrict__ cnt, int total) {
  int t = blockIdx.x * blockDim.x + threadIdx.x;
  if (t >= total) return;
  int b = t / EEE, e = t % EEE;
  int dst = fedges[(size_t)b * 2 * EEE + EEE + e] + b * NFF;
  atomicAdd(&cnt[dst], 1.0f);
}

// edge aggregation: read bf16 features, accumulate f32 atomics
__global__ void agg_kernel(const int* __restrict__ fedges, const __bf16* __restrict__ x,
                           float* __restrict__ agg, int total) {
  int t = blockIdx.x * blockDim.x + threadIdx.x;
  if (t >= total) return;
  int ch = t & 63;
  int te = t >> 6;
  int b = te / EEE, e = te % EEE;
  int src = fedges[(size_t)b * 2 * EEE + e] + b * NFF;
  int dst = fedges[(size_t)b * 2 * EEE + EEE + e] + b * NFF;
  const __bf16* xs = x + (size_t)src * DIMF + ch * 8;
  float* ad = agg + (size_t)dst * DIMF + ch * 8;
  v8bf xv = *(const v8bf*)xs;
#pragma unroll
  for (int i = 0; i < 8; ++i) atomicAdd(&ad[i], (float)xv[i]);
}

// mean = agg/cnt, emitted directly as bf16 GEMM operand
__global__ void meanb_kernel(const float* __restrict__ agg, const float* __restrict__ cnt,
                             __bf16* __restrict__ aggb, int total) {
  int i = blockIdx.x * blockDim.x + threadIdx.x;
  if (i < total) aggb[i] = f2bf(agg[i] / fmaxf(cnt[i >> 9], 1.0f));   // DIMF = 512
}

__global__ void den_kernel(const int* __restrict__ faces, float* __restrict__ den, int total) {
  int t = blockIdx.x * blockDim.x + threadIdx.x;
  if (t >= total) return;
  int fr = t / 3;
  int b = fr / NFF;
  int v = faces[t];
  atomicAdd(&den[b * NVV + v], 1.0f);
}

__global__ void avgnum_kernel(const int* __restrict__ faces, const float* __restrict__ fe,
                              float* __restrict__ avg, int total) {
  int i = blockIdx.x * blockDim.x + threadIdx.x;
  if (i >= total) return;
  int d = i % DCB;
  int t = i / DCB;          // fr*3 + j
  int j = t % 3, fr = t / 3;
  int b = fr / NFF;
  int v = faces[(size_t)fr * 3 + j];
  float val = fe[(size_t)fr * (3 * DCB) + j * DCB + d];
  atomicAdd(&avg[((size_t)b * NVV + v) * DCB + d], val);
}

__global__ void avgdiv_kernel(float* __restrict__ avg, const float* __restrict__ den, int total) {
  int i = blockIdx.x * blockDim.x + threadIdx.x;
  if (i < total) avg[i] /= fmaxf(den[i / DCB], 1e-5f);
}

__global__ void out_kernel(const int* __restrict__ faces, const int* __restrict__ idx0,
                           const int* __restrict__ idx1, const float* __restrict__ cbf,
                           float* __restrict__ out, int total) {
  int i = blockIdx.x * blockDim.x + threadIdx.x;
  if (i >= total) return;
  int d = i % DCB;
  int t = i / DCB;          // fr*3 + j
  int j = t % 3, fr = t / 3;
  int b = fr / NFF;
  int v = faces[(size_t)fr * 3 + j];
  int bv = b * NVV + v;
  out[i] = cbf[(size_t)idx0[bv] * DCB + d] + cbf[(size_t)idx1[bv] * DCB + d];
}

// ---------------------------------------------------------------------------
extern "C" void kernel_launch(void* const* d_in, const int* in_sizes, int n_in,
                              void* d_out, int out_size, void* d_ws, size_t ws_size,
                              hipStream_t stream) {
  (void)in_sizes; (void)n_in; (void)out_size; (void)ws_size;
  const float* vertices   = (const float*)d_in[0];
  const int*   faces      = (const int*)  d_in[1];
  const int*   fedges     = (const int*)  d_in[2];
  const float* coor_embed = (const float*)d_in[3];
  const float* W_in = (const float*)d_in[4];   const float* b_in = (const float*)d_in[5];
  const float* Wl0  = (const float*)d_in[6];   const float* bl0  = (const float*)d_in[7];
  const float* Wr0  = (const float*)d_in[8];
  const float* Wl1  = (const float*)d_in[9];   const float* bl1  = (const float*)d_in[10];
  const float* Wr1  = (const float*)d_in[11];
  const float* Wcb  = (const float*)d_in[12];  const float* bcb  = (const float*)d_in[13];
  const float* cbf  = (const float*)d_in[14];
  float* out = (float*)d_out;

  char* ws = (char*)d_ws;
  size_t off = 0;
  auto alloc = [&](size_t bytes) -> char* {
    off = (off + 255) & ~(size_t)255;
    char* p = ws + off; off += bytes; return p;
  };
  __bf16* winT = (__bf16*)alloc((size_t)KIN  * DIMF * 2);
  __bf16* wl0T = (__bf16*)alloc((size_t)DIMF * DIMF * 2);
  __bf16* wr0T = (__bf16*)alloc((size_t)DIMF * DIMF * 2);
  __bf16* wl1T = (__bf16*)alloc((size_t)DIMF * DIMF * 2);
  __bf16* wr1T = (__bf16*)alloc((size_t)DIMF * DIMF * 2);
  __bf16* wcbT = (__bf16*)alloc((size_t)DIMF * KIN  * 2);
  __bf16* cbb  = (__bf16*)alloc((size_t)NCB * DCB * 2);
  float*  cbsq = (float*) alloc((size_t)NCB * 4);
  float*  cnt  = (float*) alloc((size_t)MROWS * 4);
  float*  den  = (float*) alloc((size_t)VROWS * 4);
  int*    idx0 = (int*)   alloc((size_t)VROWS * 4);
  int*    idx1 = (int*)   alloc((size_t)VROWS * 4);
  float*  avg  = (float*) alloc((size_t)VROWS * DCB * 4);
  // 92 MB region reused three times: EMBb(bf16) -> AGG(f32) -> FE(f32)
  char*   R    = alloc((size_t)MROWS * (3 * DCB) * 4);
  __bf16* Xb   = (__bf16*)alloc((size_t)MROWS * DIMF * 2);
  __bf16* Yb   = (__bf16*)alloc((size_t)MROWS * DIMF * 2);
  __bf16* AGGb = (__bf16*)alloc((size_t)MROWS * DIMF * 2);
  __bf16* EMBb = (__bf16*)R;
  float*  AGG  = (float*)R;
  float*  FE   = (float*)R;

  const int TPB = 256;
#define GRID(n) (((n) + TPB - 1) / TPB)
  auto gemm = [&](const __bf16* A1, const __bf16* B1, int K1,
                  const __bf16* A2, const __bf16* B2, int K2,
                  const float* bias, float* Cf, __bf16* Cb, int M, int N) {
    dim3 grid((N + 255) / 256, (M + 31) / 32);
    gemm_kernel<<<grid, dim3(256), 0, stream>>>(A1, B1, K1, A2, B2, K2, bias, Cf, Cb, M, N);
  };

  // --- prep: weight transposes (N-major bf16), codebook bf16 + squared norms
  transpose_bf16_kernel<<<GRID(KIN * DIMF),  TPB, 0, stream>>>(W_in, winT, KIN,  DIMF, KIN * DIMF);
  transpose_bf16_kernel<<<GRID(DIMF * DIMF), TPB, 0, stream>>>(Wl0,  wl0T, DIMF, DIMF, DIMF * DIMF);
  transpose_bf16_kernel<<<GRID(DIMF * DIMF), TPB, 0, stream>>>(Wr0,  wr0T, DIMF, DIMF, DIMF * DIMF);
  transpose_bf16_kernel<<<GRID(DIMF * DIMF), TPB, 0, stream>>>(Wl1,  wl1T, DIMF, DIMF, DIMF * DIMF);
  transpose_bf16_kernel<<<GRID(DIMF * DIMF), TPB, 0, stream>>>(Wr1,  wr1T, DIMF, DIMF, DIMF * DIMF);
  transpose_bf16_kernel<<<GRID(DIMF * KIN),  TPB, 0, stream>>>(Wcb,  wcbT, DIMF, KIN,  DIMF * KIN);
  cbconv_kernel<<<GRID(NCB * DCB), TPB, 0, stream>>>(cbf, cbb, NCB * DCB);
  cbsq_kernel<<<GRID(NCB), TPB, 0, stream>>>(cbf, cbsq, NCB);

  // --- encode: discretize + gather embeddings (bf16), x = emb @ W_in + b_in
  embed_kernel<<<GRID(MROWS * KIN), TPB, 0, stream>>>(vertices, faces, coor_embed, EMBb, MROWS * KIN);
  gemm(EMBb, winT, KIN, nullptr, nullptr, 0, b_in, nullptr, Xb, MROWS, DIMF);

  // --- SAGE conv 0:  Y = X@Wr0 + mean@Wl0 + bl0  (one fused dual-K GEMM)
  zero_kernel<<<GRID(MROWS), TPB, 0, stream>>>(cnt, MROWS);
  cnt_kernel<<<GRID(BB * EEE), TPB, 0, stream>>>(fedges, cnt, BB * EEE);
  zero_kernel<<<GRID(MROWS * DIMF), TPB, 0, stream>>>(AGG, MROWS * DIMF);
  agg_kernel<<<GRID(BB * EEE * 64), TPB, 0, stream>>>(fedges, Xb, AGG, BB * EEE * 64);
  meanb_kernel<<<GRID(MROWS * DIMF), TPB, 0, stream>>>(AGG, cnt, AGGb, MROWS * DIMF);
  gemm(Xb, wr0T, DIMF, AGGb, wl0T, DIMF, bl0, nullptr, Yb, MROWS, DIMF);

  // --- SAGE conv 1:  X = Y@Wr1 + mean@Wl1 + bl1
  zero_kernel<<<GRID(MROWS * DIMF), TPB, 0, stream>>>(AGG, MROWS * DIMF);
  agg_kernel<<<GRID(BB * EEE * 64), TPB, 0, stream>>>(fedges, Yb, AGG, BB * EEE * 64);
  meanb_kernel<<<GRID(MROWS * DIMF), TPB, 0, stream>>>(AGG, cnt, AGGb, MROWS * DIMF);
  gemm(Yb, wr1T, DIMF, AGGb, wl1T, DIMF, bl1, nullptr, Xb, MROWS, DIMF);

  // --- face-vertex embeds + per-vertex average
  gemm(Xb, wcbT, DIMF, nullptr, nullptr, 0, bcb, FE, nullptr, MROWS, 3 * DCB);
  zero_kernel<<<GRID(VROWS * DCB), TPB, 0, stream>>>(avg, VROWS * DCB);
  zero_kernel<<<GRID(VROWS), TPB, 0, stream>>>(den, VROWS);
  den_kernel<<<GRID(BB * NFF * 3), TPB, 0, stream>>>(faces, den, BB * NFF * 3);
  avgnum_kernel<<<GRID(BB * NFF * 3 * DCB), TPB, 0, stream>>>(faces, FE, avg, BB * NFF * 3 * DCB);
  avgdiv_kernel<<<GRID(VROWS * DCB), TPB, 0, stream>>>(avg, den, VROWS * DCB);

  // --- residual VQ (2 quantizers fused in one WMMA kernel) + output gather
  vq_kernel<<<dim3(VROWS / 16), dim3(32), 0, stream>>>(avg, cbb, cbf, cbsq, idx0, idx1);
  out_kernel<<<GRID(BB * NFF * 3 * DCB), TPB, 0, stream>>>(faces, idx0, idx1, cbf, out,
                                                           BB * NFF * 3 * DCB);
#undef GRID
}
